// AffineTransform_35914516529617
// MI455X (gfx1250) — matrix operand order repaired and checked by reading, not compiled, not run on hardware
//
#include <hip/hip_runtime.h>
#include <hip/hip_bf16.h>

// ---- problem constants -------------------------------------------------
#define BSZ 2048
#define DDIM 256
#define ZDIM 512
#define ADIM 64
#define NLAYER 3

// ---- GEMM tiling -------------------------------------------------------
#define BM 32
#define BN 128
#define OUT_STRIDE 260   // 32x256 out block, padded (260%64==4 -> conflict-free col gathers)
#define S_STRIDE   65    // 32x64  s block, padded

typedef __attribute__((ext_vector_type(16))) __bf16 v16bf;
typedef __attribute__((ext_vector_type(8)))  float  v8f;

__device__ __forceinline__ v8f wmma_bf16(v16bf a, v16bf b, v8f c) {
    // (neg_a, A, neg_b, B, c_mod, C, reuse_a, reuse_b)
    return __builtin_amdgcn_wmma_f32_16x16x32_bf16(false, a, false, b, (short)0, c,
                                                   false, false);
}

union FragU {
    uint4  q[2];
    v16bf  v;
};

// ---- router: s = softmax(relu(x@w1^T+b1)@w2^T+b2), one block per row ----
__global__ __launch_bounds__(256)
void router_kernel(const float* __restrict__ x,
                   const float* __restrict__ w1, const float* __restrict__ b1,
                   const float* __restrict__ w2, const float* __restrict__ b2,
                   float* __restrict__ s_out) {
    __shared__ float xrow[DDIM];
    __shared__ float hrow[ZDIM];
    __shared__ float lgts[ADIM];
    __shared__ float red[2];
    const int t = threadIdx.x;
    const int b = blockIdx.x;

    xrow[t] = x[b * DDIM + t];
    __syncthreads();

    for (int z = t; z < ZDIM; z += 256) {
        float acc = b1[z];
        const float* wr = w1 + (size_t)z * DDIM;
        #pragma unroll 8
        for (int k = 0; k < DDIM; ++k) acc += xrow[k] * wr[k];
        hrow[z] = acc > 0.0f ? acc : 0.0f;
    }
    __syncthreads();

    if (t < ADIM) {
        float acc = b2[t];
        const float* wr = w2 + (size_t)t * ZDIM;
        #pragma unroll 8
        for (int k = 0; k < ZDIM; ++k) acc += hrow[k] * wr[k];
        lgts[t] = acc;
    }
    __syncthreads();

    if (t == 0) {
        float m = lgts[0];
        for (int a = 1; a < ADIM; ++a) m = fmaxf(m, lgts[a]);
        float sum = 0.0f;
        for (int a = 0; a < ADIM; ++a) sum += __expf(lgts[a] - m);
        red[0] = m;
        red[1] = 1.0f / sum;
    }
    __syncthreads();

    if (t < ADIM)
        s_out[b * ADIM + t] = __expf(lgts[t] - red[0]) * red[1];
}

// ---- off = s @ AO_layer  ([2048x64]@[64x256]) ---------------------------
__global__ __launch_bounds__(256)
void offset_kernel(const float* __restrict__ s, const float* __restrict__ AO,
                   float* __restrict__ off) {
    const int idx = blockIdx.x * 256 + threadIdx.x;  // one block == one row b
    const int b = idx >> 8;
    const int i = idx & 255;
    float acc = 0.0f;
    #pragma unroll 8
    for (int a = 0; a < ADIM; ++a)
        acc += s[b * ADIM + a] * AO[a * DDIM + i];
    off[idx] = acc;
}

// ---- pack W = AM^T(a-blockwise) into bf16 hi/lo in WMMA-B fragment order
// Bpk[k16][n][kk] with k16 = a*16 + j/16, kk = j%16  (K contiguous per lane)
// stored as uint pairs: whi/wlo[ (k16*256 + n)*8 + (j%16)/2 ]
__global__ __launch_bounds__(256)
void pack_w_kernel(const float* __restrict__ AMl,          // [64][65536]
                   unsigned int* __restrict__ whi,
                   unsigned int* __restrict__ wlo) {
    const int id = blockIdx.x * 256 + threadIdx.x;         // 64*256*128
    const int jp = id & 127;
    const int n  = (id >> 7) & 255;
    const int a  = id >> 15;
    const int j  = jp << 1;

    const size_t src = ((size_t)a << 16) + (size_t)n * DDIM + j;
    const float x0 = AMl[src];
    const float x1 = AMl[src + 1];
    const __bf16 h0 = (__bf16)x0, h1 = (__bf16)x1;
    const __bf16 l0 = (__bf16)(x0 - (float)h0);
    const __bf16 l1 = (__bf16)(x1 - (float)h1);

    const int k16 = (a << 4) + (j >> 4);
    const size_t o = ((size_t)k16 * DDIM + n) * 8 + ((j >> 1) & 7);

    union { __bf16 b[2]; unsigned int u; } ph, pl;
    ph.b[0] = h0; ph.b[1] = h1;
    pl.b[0] = l0; pl.b[1] = l1;
    whi[o] = ph.u;
    wlo[o] = pl.u;
}

// ---- main layer GEMM: out' = prelu?( T(s,out) @ W + off ) ---------------
// C[2048x256] = T[2048x16384] x W[16384x256]
//   T[b, a*256+j] = s[b,a]*out[b,j]   built in registers from LDS (split bf16)
//   W pre-split/pre-swizzled in whi/wlo -> direct global_load_b128 fragments,
//   single per-lane pointer advanced by a constant 16 KB per k-slice
// split-bf16: C += Ahi*Bhi + Alo*Bhi + Ahi*Blo  (fp32 accumulators)
__global__ __launch_bounds__(256, 1)
void layer_gemm_kernel(const float* __restrict__ out_prev,
                       const float* __restrict__ s,
                       const unsigned int* __restrict__ whi,
                       const unsigned int* __restrict__ wlo,
                       const float* __restrict__ off,
                       const float* __restrict__ prelu_a,
                       float* __restrict__ out_next,
                       int apply_prelu) {
    __shared__ float lds_out[BM * OUT_STRIDE];  // 33.3 KB
    __shared__ float lds_s[BM * S_STRIDE];      //  8.3 KB

    const int t  = threadIdx.x;
    const int b0 = blockIdx.x * BM;
    const int n0 = blockIdx.y * BN;

    // stage out block (32x256) and s block (32x64), coalesced
    #pragma unroll
    for (int e = 0; e < 32; ++e) {
        int idx = t + e * 256;
        lds_out[(idx >> 8) * OUT_STRIDE + (idx & 255)] =
            out_prev[(size_t)(b0 + (idx >> 8)) * DDIM + (idx & 255)];
    }
    #pragma unroll
    for (int e = 0; e < 8; ++e) {
        int idx = t + e * 256;
        lds_s[(idx >> 6) * S_STRIDE + (idx & 63)] =
            s[(size_t)(b0 + (idx >> 6)) * ADIM + (idx & 63)];
    }
    __syncthreads();

    const int lane = t & 31;
    const int wave = t >> 5;
    const int hv   = lane >> 4;   // lane half
    const int l16  = lane & 15;
    const int tileRow  = wave >> 2;          // 0..1 (16 rows each)
    const int c0       = (wave & 3) * 2;     // 2 tile-cols per wave
    const int rowLocal = tileRow * 16 + l16;

    const int gn0 = n0 + c0 * 16 + l16;      // global n of tile 0 (tile 1 = +16)

    // per-lane streaming pointers over packed W; k16 = a*16 + kj/16 + hv.
    // advance per kj slice: 2*256*8 uints = 4096; 8 slices == one 'a' step.
    const unsigned int* pHi = whi + ((size_t)hv * DDIM + gn0) * 8;
    const unsigned int* pLo = wlo + ((size_t)hv * DDIM + gn0) * 8;
    const float* pOut = lds_out + rowLocal * OUT_STRIDE;

    v8f acc0 = {};
    v8f acc1 = {};

    for (int a = 0; a < ADIM; ++a) {
        const float sval = lds_s[rowLocal * S_STRIDE + a];

        #pragma unroll 2
        for (int kj = 0; kj < DDIM; kj += 32) {
            // ---- A fragment (T = s*out), A-layout: K=half*16+v*2
            v16bf ahi, alo;
            #pragma unroll
            for (int v = 0; v < 8; ++v) {
                int kb = (hv << 4) + (v << 1);
                float p0 = sval * pOut[kj + kb];
                float p1 = sval * pOut[kj + kb + 1];
                __bf16 h0 = (__bf16)p0, h1 = (__bf16)p1;
                ahi[2 * v]     = h0;
                ahi[2 * v + 1] = h1;
                alo[2 * v]     = (__bf16)(p0 - (float)h0);
                alo[2 * v + 1] = (__bf16)(p1 - (float)h1);
            }

            // ---- B fragments: coalesced b128 loads; tile1 via +512B imm offset
            const uint4* qh = (const uint4*)pHi;
            const uint4* ql = (const uint4*)pLo;
            FragU bh0, bl0, bh1, bl1;
            bh0.q[0] = qh[0];  bh0.q[1] = qh[1];    // tile0 hi  (+0, +16B)
            bl0.q[0] = ql[0];  bl0.q[1] = ql[1];    // tile0 lo
            bh1.q[0] = qh[32]; bh1.q[1] = qh[33];   // tile1 hi  (+512B, +528B)
            bl1.q[0] = ql[32]; bl1.q[1] = ql[33];   // tile1 lo

            acc0 = wmma_bf16(ahi, bh0.v, acc0);
            acc0 = wmma_bf16(alo, bh0.v, acc0);
            acc0 = wmma_bf16(ahi, bl0.v, acc0);
            acc1 = wmma_bf16(ahi, bh1.v, acc1);
            acc1 = wmma_bf16(alo, bh1.v, acc1);
            acc1 = wmma_bf16(ahi, bl1.v, acc1);

            pHi += 4096;   // next k-slice (16 KB)
            pLo += 4096;
        }
    }

    // ---- epilogue: + off, optional PReLU; C layout: M = r + half*8, N = l16
    const float pa = *prelu_a;
    #pragma unroll
    for (int ct = 0; ct < 2; ++ct) {
        #pragma unroll
        for (int r = 0; r < 8; ++r) {
            int gb = b0 + tileRow * 16 + (hv << 3) + r;
            int gn = gn0 + ct * 16;
            float v = (ct == 0 ? acc0[r] : acc1[r]) + off[(size_t)gb * DDIM + gn];
            if (apply_prelu) v = v >= 0.0f ? v : pa * v;
            out_next[(size_t)gb * DDIM + gn] = v;
        }
    }
}

extern "C" void kernel_launch(void* const* d_in, const int* in_sizes, int n_in,
                              void* d_out, int out_size, void* d_ws, size_t ws_size,
                              hipStream_t stream) {
    const float* x  = (const float*)d_in[0];
    const float* AM = (const float*)d_in[1];   // [3][64][65536]
    const float* AO = (const float*)d_in[2];   // [3][64][256]
    const float* w1 = (const float*)d_in[3];
    const float* b1 = (const float*)d_in[4];
    const float* w2 = (const float*)d_in[5];
    const float* b2 = (const float*)d_in[6];
    const float* pa = (const float*)d_in[7];
    float* out = (float*)d_out;

    float* ws   = (float*)d_ws;
    float* sbuf = ws;                          // 2048*64
    float* offb = sbuf + BSZ * ADIM;           // 2048*256
    float* bufA = offb + BSZ * DDIM;           // 2048*256
    float* bufB = bufA + BSZ * DDIM;           // 2048*256
    unsigned int* whi = (unsigned int*)(bufB + BSZ * DDIM);      // 2M uints (8MB)
    unsigned int* wlo = whi + (size_t)ADIM * DDIM * DDIM / 2;    // 2M uints (8MB)

    router_kernel<<<BSZ, 256, 0, stream>>>(x, w1, b1, w2, b2, sbuf);

    const float* cur = x;
    float* nxt[NLAYER] = {bufA, bufB, out};
    for (int l = 0; l < NLAYER; ++l) {
        pack_w_kernel<<<(ADIM * DDIM * DDIM / 2) / 256, 256, 0, stream>>>(
            AM + (size_t)l * ADIM * DDIM * DDIM, whi, wlo);
        offset_kernel<<<(BSZ * DDIM) / 256, 256, 0, stream>>>(
            sbuf, AO + (size_t)l * ADIM * DDIM, offb);
        dim3 grid(BSZ / BM, DDIM / BN);
        layer_gemm_kernel<<<grid, 256, 0, stream>>>(
            cur, sbuf, whi, wlo, offb, pa,
            nxt[l], (l < NLAYER - 1) ? 1 : 0);
        cur = nxt[l];
    }
}
